// RNNModel_90151363543078
// MI455X (gfx1250) — matrix operand order repaired and checked by reading, not compiled, not run on hardware
//
#include <hip/hip_runtime.h>

// ---------------------------------------------------------------------------
// RNN model: dense head -> GRU1(U1=512) -> GRU2(U2=256) -> sigmoid dot head
// B=128, T=1024, F=64. Fused persistent-recurrence design:
//   8 workgroups x 512 threads (16 waves), each owns 16 batch rows for all
//   1024 steps. GEMMs via v_wmma_f32_16x16x32_bf16. A-matrix fragments are
//   hoisted into VGPRs once per step per wave (LDS-bandwidth relief); B
//   streams from global (L2/WGP$-resident bf16 transposed weights).
// ---------------------------------------------------------------------------

typedef __bf16 bf16_t;
typedef __attribute__((ext_vector_type(16))) __bf16 v16bf;
typedef __attribute__((ext_vector_type(8)))  __bf16 v8bf;
typedef __attribute__((ext_vector_type(8)))  float  v8f;

#define Bsz  128
#define Tsz  1024
#define Fsz  64
#define U1   512
#define U2   256
#define DU   256

#define WAVES   16
#define THREADS (WAVES * 32)
#define NBLK    (Bsz / 16)

// ---- workspace byte offsets (all 32B aligned) ----
#define OFF_K1T   0u                           // bf16 [1536][64]
#define OFF_RK1T  196608u                      // bf16 [1536][512]
#define OFF_K2T   1769472u                     // bf16 [768][512]
#define OFF_RK2T  2555904u                     // bf16 [768][256]
#define OFF_TMP   2949120u                     // f32  [128][256]
#define OFF_H0    3080192u                     // f32  [128][512]

// ---- LDS row strides (elements), padded for bank spread ----
#define SX   72     // xbuf   bf16 [16][72]
#define S1B  520    // h1bf   bf16 [16][520]
#define S2B  264    // h2bf   bf16 [16][264]
#define S1F  516    // h1f    f32  [16][516]
#define SZR  1032   // zrbuf  f32  [16][1032]
#define SXH  516    // xh/hh  f32  [16][516]
#define S2F  260    // h2f    f32  [16][260]
#define SZ2  520    // zr2    f32  [16][520]
#define SX2  260    // xh2/hh2 f32 [16][260]

// ---- LDS byte offsets ----
#define L_XBUF   0u
#define L_H1B    2304u
#define L_H2B    18944u
#define L_H1F    27392u
#define L_ZR     60416u
#define L_XH     126464u
#define L_HH     159488u
#define L_H2F    192512u
#define L_ZR2    209152u
#define L_XH2    242432u
#define L_HH2    259072u
#define L_RED    275712u
#define LDS_BYTES 276736u

__device__ __forceinline__ float sigmoidf_(float x) { return 1.0f / (1.0f + __expf(-x)); }
__device__ __forceinline__ float eluf_(float x)     { return x > 0.0f ? x : (__expf(x) - 1.0f); }

__device__ __forceinline__ v8f wmma_bf16(v16bf a, v16bf b, v8f c) {
  // D = A(16x32 bf16) * B(32x16 bf16) + C(16x16 f32)
  return __builtin_amdgcn_wmma_f32_16x16x32_bf16(false, a, false, b, (short)0, c, false, false);
}

// A fragment (16x32, bf16, row-major in LDS): lane m(0..15) holds K=kb..kb+7
// and kb+16..kb+23 with kb = k0 + 8*(lane>>4)  [ISA 16-bit A layout]
__device__ __forceinline__ v16bf load_a_frag(const bf16_t* base, int stride, int lane, int k0) {
  const int m  = lane & 15;
  const int kb = k0 + ((lane >> 4) << 3);
  const bf16_t* p = base + m * stride + kb;
  v8bf lo = *(const v8bf*)(p);
  v8bf hi = *(const v8bf*)(p + 16);
  return __builtin_shufflevector(lo, hi, 0, 1, 2, 3, 4, 5, 6, 7, 8, 9, 10, 11, 12, 13, 14, 15);
}

// B fragment (32x16, bf16): weight stored transposed W^T[N][K] row-major in
// global; lane n(0..15) holds K=kb..kb+7 and kb+16..kb+23, kb = k0 + 8*(lane>>4).
__device__ __forceinline__ v16bf load_b_frag(const bf16_t* wt, int K, int n0, int lane, int k0) {
  const int n  = n0 + (lane & 15);
  const int kb = k0 + ((lane >> 4) << 3);
  const bf16_t* p = wt + (size_t)n * K + kb;
  v8bf lo = *(const v8bf*)(p);
  v8bf hi = *(const v8bf*)(p + 16);
  return __builtin_shufflevector(lo, hi, 0, 1, 2, 3, 4, 5, 6, 7, 8, 9, 10, 11, 12, 13, 14, 15);
}

// ---------------------------------------------------------------------------
// Weight transpose + f32->bf16 convert:  in f32 [K][N] -> out bf16 [N][K]
// ---------------------------------------------------------------------------
__global__ void convT_kernel(const float* __restrict__ in, bf16_t* __restrict__ out,
                             int K, int N) {
  int i = blockIdx.x * blockDim.x + threadIdx.x;
  if (i >= K * N) return;
  int k = i / N, n = i - k * N;
  out[(size_t)n * K + k] = (bf16_t)in[i];
}

// ---------------------------------------------------------------------------
// Dense head (tiny): tmp = input1 @ w_h1 + b_h1 ; h0 = tanh(tmp @ w_out + b_out)
// ---------------------------------------------------------------------------
__global__ void head1_kernel(const float* __restrict__ in1, const float* __restrict__ w,
                             const float* __restrict__ b, float* __restrict__ tmp) {
  int i = blockIdx.x * blockDim.x + threadIdx.x;
  if (i >= Bsz * DU) return;
  int r = i >> 8, c = i & (DU - 1);
  float s = b[c];
  #pragma unroll 8
  for (int k = 0; k < Fsz; ++k) s += in1[r * Fsz + k] * w[k * DU + c];
  tmp[i] = s;
}

__global__ void head2_kernel(const float* __restrict__ tmp, const float* __restrict__ w,
                             const float* __restrict__ b, float* __restrict__ h0) {
  int i = blockIdx.x * blockDim.x + threadIdx.x;
  if (i >= Bsz * U1) return;
  int r = i >> 9, c = i & (U1 - 1);
  float s = b[c];
  #pragma unroll 8
  for (int k = 0; k < DU; ++k) s += tmp[r * DU + k] * w[k * U1 + c];
  h0[i] = tanhf(s);
}

// ---------------------------------------------------------------------------
// Fused persistent GRU1+GRU2+output. One block = 16 batch rows, all T steps.
// ---------------------------------------------------------------------------
__global__ __launch_bounds__(THREADS, 1)
void gru_fused_kernel(const float* __restrict__ input2,       // [B,T,F]
                      const unsigned char* __restrict__ mask, // [B,T]
                      const float* __restrict__ h0,           // [B,U1]
                      const bf16_t* __restrict__ k1t,         // [1536][64]
                      const bf16_t* __restrict__ rk1t,        // [1536][512]
                      const bf16_t* __restrict__ k2t,         // [768][512]
                      const bf16_t* __restrict__ rk2t,        // [768][256]
                      const float* __restrict__ bias1,        // [2][1536]
                      const float* __restrict__ bias2,        // [2][768]
                      const float* __restrict__ w_d,          // [256]
                      const float* __restrict__ b_d,          // [1]
                      float* __restrict__ out)                // [B,T]
{
  extern __shared__ char smem[];
  bf16_t* xbuf = (bf16_t*)(smem + L_XBUF);
  bf16_t* h1b  = (bf16_t*)(smem + L_H1B);
  bf16_t* h2b  = (bf16_t*)(smem + L_H2B);
  float*  h1f  = (float*) (smem + L_H1F);
  float*  zr   = (float*) (smem + L_ZR);
  float*  xh   = (float*) (smem + L_XH);
  float*  hh   = (float*) (smem + L_HH);
  float*  h2f  = (float*) (smem + L_H2F);
  float*  zr2  = (float*) (smem + L_ZR2);
  float*  xh2  = (float*) (smem + L_XH2);
  float*  hh2  = (float*) (smem + L_HH2);
  float*  red  = (float*) (smem + L_RED);

  const int b0   = blockIdx.x * 16;
  const int tid  = threadIdx.x;
  const int lane = tid & 31;
  const int wv   = tid >> 5;
  const int row0 = (lane >> 4) << 3;   // C-fragment row base per lane half
  const int ncol = lane & 15;          // C-fragment column within tile

  // init hidden states
  for (int i = tid; i < 16 * U1; i += THREADS) {
    int r = i >> 9, c = i & (U1 - 1);
    float v = h0[(size_t)(b0 + r) * U1 + c];
    h1f[r * S1F + c] = v;
    h1b[r * S1B + c] = (bf16_t)v;
  }
  for (int i = tid; i < 16 * U2; i += THREADS) {
    int r = i >> 8, c = i & (U2 - 1);
    h2f[r * S2F + c] = 0.0f;
    h2b[r * S2B + c] = (bf16_t)0.0f;
  }
  __syncthreads();

  const float bd = b_d[0];

  for (int t = 0; t < Tsz; ++t) {
    // ---- stage x_t [16,64] -> LDS bf16 ----
    for (int i = tid; i < 16 * Fsz; i += THREADS) {
      int r = i >> 6, c = i & (Fsz - 1);
      xbuf[r * SX + c] = (bf16_t)input2[((size_t)(b0 + r) * Tsz + t) * Fsz + c];
    }
    __syncthreads();

    // ---- GRU1: hoist shared A fragments into VGPRs (once per step) ----
    {
      v16bf ax[2];
      #pragma unroll
      for (int i = 0; i < 2; ++i) ax[i] = load_a_frag(xbuf, SX, lane, i * 32);
      v16bf ah[16];
      #pragma unroll
      for (int i = 0; i < 16; ++i) ah[i] = load_a_frag(h1b, S1B, lane, i * 32);

      // 96 tiles of 16 cols over N=1536; B streams from global only
      for (int tile = wv; tile < 96; tile += WAVES) {
        if (tile < 64) {
          // z/r region: fuse x@k1 + h1@rk1 + bias0 + bias1
          const int n0 = tile * 16;
          v8f acc = {};
          #pragma unroll
          for (int i = 0; i < 2; ++i)
            acc = wmma_bf16(ax[i], load_b_frag(k1t, Fsz, n0, lane, i * 32), acc);
          #pragma unroll
          for (int i = 0; i < 16; ++i)
            acc = wmma_bf16(ah[i], load_b_frag(rk1t, U1, n0, lane, i * 32), acc);
          const int col = n0 + ncol;
          const float badd = bias1[col] + bias1[1536 + col];
          #pragma unroll
          for (int v = 0; v < 8; ++v)
            zr[(row0 + v) * SZR + col] = acc[v] + badd;
        } else {
          // h region: keep x-part and h-part separate (cand needs r*hh)
          const int n0h = (tile - 64) * 16;
          const int nc  = 1024 + n0h;       // column in the 1536-wide space
          v8f xacc = {}, hacc = {};
          #pragma unroll
          for (int i = 0; i < 2; ++i)
            xacc = wmma_bf16(ax[i], load_b_frag(k1t, Fsz, nc, lane, i * 32), xacc);
          #pragma unroll
          for (int i = 0; i < 16; ++i)
            hacc = wmma_bf16(ah[i], load_b_frag(rk1t, U1, nc, lane, i * 32), hacc);
          const int col = n0h + ncol;
          const float bx = bias1[1024 + col];
          const float bh = bias1[1536 + 1024 + col];
          #pragma unroll
          for (int v = 0; v < 8; ++v) {
            xh[(row0 + v) * SXH + col] = xacc[v] + bx;
            hh[(row0 + v) * SXH + col] = hacc[v] + bh;
          }
        }
      }
    }
    __syncthreads();

    // ---- GRU1 gates (16x512 elements) ----
    for (int i = tid; i < 16 * U1; i += THREADS) {
      int r = i >> 9, c = i & (U1 - 1);
      float z  = sigmoidf_(zr[r * SZR + c]);
      float rr = sigmoidf_(zr[r * SZR + U1 + c]);
      float cand = eluf_(xh[r * SXH + c] + rr * hh[r * SXH + c]);
      float hp = h1f[r * S1F + c];
      float hn = z * hp + (1.0f - z) * cand;
      if (!mask[(size_t)(b0 + r) * Tsz + t]) hn = hp;
      h1f[r * S1F + c] = hn;
      h1b[r * S1B + c] = (bf16_t)hn;
    }
    __syncthreads();

    // ---- GRU2: hoist fresh-h1 A fragments; 48 tiles over N=768 ----
    {
      v16bf ah[16];
      #pragma unroll
      for (int i = 0; i < 16; ++i) ah[i] = load_a_frag(h1b, S1B, lane, i * 32);

      for (int tile = wv; tile < 48; tile += WAVES) {
        if (tile < 32) {
          const int n0 = tile * 16;
          v8f acc = {};
          #pragma unroll
          for (int i = 0; i < 16; ++i)
            acc = wmma_bf16(ah[i], load_b_frag(k2t, U1, n0, lane, i * 32), acc);
          #pragma unroll
          for (int i = 0; i < 8; ++i)
            acc = wmma_bf16(load_a_frag(h2b, S2B, lane, i * 32),
                            load_b_frag(rk2t, U2, n0, lane, i * 32), acc);
          const int col = n0 + ncol;
          const float badd = bias2[col] + bias2[768 + col];
          #pragma unroll
          for (int v = 0; v < 8; ++v)
            zr2[(row0 + v) * SZ2 + col] = acc[v] + badd;
        } else {
          const int n0h = (tile - 32) * 16;
          const int nc  = 512 + n0h;
          v8f xacc = {}, hacc = {};
          #pragma unroll
          for (int i = 0; i < 16; ++i)
            xacc = wmma_bf16(ah[i], load_b_frag(k2t, U1, nc, lane, i * 32), xacc);
          #pragma unroll
          for (int i = 0; i < 8; ++i)
            hacc = wmma_bf16(load_a_frag(h2b, S2B, lane, i * 32),
                             load_b_frag(rk2t, U2, nc, lane, i * 32), hacc);
          const int col = n0h + ncol;
          const float bx = bias2[512 + col];
          const float bh = bias2[768 + 512 + col];
          #pragma unroll
          for (int v = 0; v < 8; ++v) {
            xh2[(row0 + v) * SX2 + col] = xacc[v] + bx;
            hh2[(row0 + v) * SX2 + col] = hacc[v] + bh;
          }
        }
      }
    }
    __syncthreads();

    // ---- GRU2 gates (16x256 elements) ----
    for (int i = tid; i < 16 * U2; i += THREADS) {
      int r = i >> 8, c = i & (U2 - 1);
      float z  = sigmoidf_(zr2[r * SZ2 + c]);
      float rr = sigmoidf_(zr2[r * SZ2 + U2 + c]);
      float cand = eluf_(xh2[r * SX2 + c] + rr * hh2[r * SX2 + c]);
      float hp = h2f[r * S2F + c];
      float hn = z * hp + (1.0f - z) * cand;
      if (!mask[(size_t)(b0 + r) * Tsz + t]) hn = hp;
      h2f[r * S2F + c] = hn;
    }
    __syncthreads();

    // ---- output head: out[b,t] = sigmoid(h2 . w_d + b_d) ----
    if (tid < 256) {
      int r = tid >> 4;
      int c0 = (tid & 15) * 16;
      float s = 0.0f;
      #pragma unroll
      for (int c = 0; c < 16; ++c) s += h2f[r * S2F + c0 + c] * w_d[c0 + c];
      red[tid] = s;
    }
    __syncthreads();
    if (tid < 16) {
      float s = 0.0f;
      #pragma unroll
      for (int j = 0; j < 16; ++j) s += red[tid * 16 + j];
      out[(size_t)(b0 + tid) * Tsz + t] = sigmoidf_(s + bd);
    }
    __syncthreads();
  }
}

// ---------------------------------------------------------------------------
extern "C" void kernel_launch(void* const* d_in, const int* in_sizes, int n_in,
                              void* d_out, int out_size, void* d_ws, size_t ws_size,
                              hipStream_t stream) {
  const float* input1 = (const float*)d_in[0];
  const float* input2 = (const float*)d_in[1];
  const unsigned char* mask = (const unsigned char*)d_in[2];
  const float* w_h1  = (const float*)d_in[3];
  const float* b_h1  = (const float*)d_in[4];
  const float* w_out = (const float*)d_in[5];
  const float* b_out = (const float*)d_in[6];
  const float* k1    = (const float*)d_in[7];
  const float* rk1   = (const float*)d_in[8];
  const float* bias1 = (const float*)d_in[9];
  const float* k2    = (const float*)d_in[10];
  const float* rk2   = (const float*)d_in[11];
  const float* bias2 = (const float*)d_in[12];
  const float* w_d   = (const float*)d_in[13];
  const float* b_d   = (const float*)d_in[14];
  float* out = (float*)d_out;

  char* ws = (char*)d_ws;
  bf16_t* k1t  = (bf16_t*)(ws + OFF_K1T);
  bf16_t* rk1t = (bf16_t*)(ws + OFF_RK1T);
  bf16_t* k2t  = (bf16_t*)(ws + OFF_K2T);
  bf16_t* rk2t = (bf16_t*)(ws + OFF_RK2T);
  float*  tmp  = (float*) (ws + OFF_TMP);
  float*  h0   = (float*) (ws + OFF_H0);

  // weight transpose+convert (one-shot, cheap; deterministic each call)
  {
    int n;
    n = Fsz * 3 * U1; convT_kernel<<<(n + 255) / 256, 256, 0, stream>>>(k1,  k1t,  Fsz, 3 * U1);
    n = U1  * 3 * U1; convT_kernel<<<(n + 255) / 256, 256, 0, stream>>>(rk1, rk1t, U1,  3 * U1);
    n = U1  * 3 * U2; convT_kernel<<<(n + 255) / 256, 256, 0, stream>>>(k2,  k2t,  U1,  3 * U2);
    n = U2  * 3 * U2; convT_kernel<<<(n + 255) / 256, 256, 0, stream>>>(rk2, rk2t, U2,  3 * U2);
  }

  // dense head -> h0
  head1_kernel<<<(Bsz * DU + 255) / 256, 256, 0, stream>>>(input1, w_h1, b_h1, tmp);
  head2_kernel<<<(Bsz * U1 + 255) / 256, 256, 0, stream>>>(tmp, w_out, b_out, h0);

  // fused persistent recurrence (needs >64KB dynamic LDS)
  (void)hipFuncSetAttribute((const void*)gru_fused_kernel,
                            hipFuncAttributeMaxDynamicSharedMemorySize,
                            (int)LDS_BYTES);
  gru_fused_kernel<<<NBLK, THREADS, LDS_BYTES, stream>>>(
      input2, mask, h0, k1t, rk1t, k2t, rk2t, bias1, bias2, w_d, b_d, out);
}
